// GPT_71906342470237
// MI455X (gfx1250) — matrix operand order repaired and checked by reading, not compiled, not run on hardware
//
#include <hip/hip_runtime.h>
#include <hip/hip_bf16.h>
#include <math.h>

// ---------------------------------------------------------------------------
// RWKV-7 forward for MI455X (gfx1250).
// All dense projections go through v_wmma_f32_16x16x32_f16 (wave32 WMMA).
// GEMMs are matrix-pipe bound (weights ~250MB stream once; L2 is 192MB), so
// fp32 activations are converted to f16 in a vectorized LDS staging path
// (global_load_b128 -> cvt -> ds_store_b128).
// The 64x64-per-head state recurrence is a register-resident scan kernel.
// ---------------------------------------------------------------------------

typedef __attribute__((ext_vector_type(16))) _Float16 v16h;
typedef __attribute__((ext_vector_type(8)))  _Float16 v8h;
typedef __attribute__((ext_vector_type(8)))  float    v8f;
typedef __attribute__((ext_vector_type(4)))  float    v4f;

#define VOCAB   50304
#define CDIM    768
#define NLAYER  6
#define HS      64
#define NH      12
#define FFN     3072
#define BBATCH  2
#define TSEQ    1024
#define MTOK    (BBATCH*TSEQ)   // 2048 tokens

#define TBM 64
#define TBN 128
#define TBK 32

// ------------------------------ epilogues ----------------------------------
// 0=none 1=tanh 2=sigmoid 3=rwkv decay exp(-exp(-softplus(-z)-0.5)) 4=relu^2
template<int EPI>
__device__ __forceinline__ float apply_epi(float z) {
  if constexpr (EPI == 1) {
    return tanhf(z);
  } else if constexpr (EPI == 2) {
    return 1.f / (1.f + __expf(-z));
  } else if constexpr (EPI == 3) {
    float w = -log1pf(__expf(-z)) - 0.5f;   // -softplus(-z) - 0.5
    return __expf(-__expf(w));              // store decay directly for scan
  } else if constexpr (EPI == 4) {
    float t = fmaxf(z, 0.f);
    return t * t;
  } else {
    return z;
  }
}

// ------------------------------ WMMA GEMM ----------------------------------
// out[M,N] = epi(bias[n] + A'[M,K] @ B[K,N]) (+ addsrc[M,N] if ADD)
// A'[m,k] = A[m,k] + AX[m,k]*mix[k] when MIX (RWKV token-mix fuse).
template<int EPI, bool MIX, bool ADD>
__global__ __launch_bounds__(256)
void gemm_wmma(const float* __restrict__ A, const float* __restrict__ AX,
               const float* __restrict__ mix, const float* __restrict__ Bm,
               const float* __restrict__ bias, const float* __restrict__ addsrc,
               float* __restrict__ out, int M, int N, int K) {
  __shared__ __align__(16) _Float16 As[TBM][TBK + 8];   // row stride 80B
  __shared__ __align__(16) _Float16 Bs[TBK][TBN + 8];   // row stride 272B

  const int tid  = threadIdx.x;
  const int m0   = blockIdx.y * TBM;
  const int n0   = blockIdx.x * TBN;
  const int wave = tid >> 5;
  const int lane = tid & 31;
  const int wm   = wave >> 2;      // 0..1  -> 32-row slab
  const int wn   = wave & 3;       // 0..3  -> 32-col slab
  const int hlf  = lane >> 4;      // lane half (ISA A/D layout split)
  const int lr   = lane & 15;

  const int arow  = tid >> 2;          // 0..63
  const int akoff = (tid & 3) * 8;     // 0,8,16,24  (16B-aligned in halves)
  const int brow  = tid >> 3;          // 0..31
  const int bcoff = (tid & 7) * 16;    // 0..112     (32B-aligned in halves)

  const bool full_n = (n0 + TBN <= N);   // uniform across block

  v8f acc[2][2] = {};

  for (int k0 = 0; k0 < K; k0 += TBK) {
    // ---- stage A tile (64x32): 2x float4 loads -> one 16B LDS store ----
    {
      const int gm = m0 + arow;
      const float* ap = A + (size_t)gm * K + k0 + akoff;
      v4f a0 = *(const v4f*)ap;
      v4f a1 = *(const v4f*)(ap + 4);
      if constexpr (MIX) {
        const float* xp = AX + (size_t)gm * K + k0 + akoff;
        const float* mp = mix + k0 + akoff;
        v4f x0 = *(const v4f*)xp;
        v4f x1 = *(const v4f*)(xp + 4);
        v4f q0 = *(const v4f*)mp;
        v4f q1 = *(const v4f*)(mp + 4);
        a0 += x0 * q0;
        a1 += x1 * q1;
      }
      v8h h;
      #pragma unroll
      for (int i = 0; i < 4; i++) {
        h[i]     = (_Float16)a0[i];
        h[i + 4] = (_Float16)a1[i];
      }
      *(v8h*)&As[arow][akoff] = h;
      if (k0 + TBK < K)
        __builtin_prefetch(ap + TBK, 0, 1);
    }
    // ---- stage B tile (32x128): 4x float4 loads -> two 16B LDS stores ----
    {
      const int gk = k0 + brow;
      if (full_n) {
        const float* bp = Bm + (size_t)gk * N + n0 + bcoff;
        v4f b0 = *(const v4f*)bp;
        v4f b1 = *(const v4f*)(bp + 4);
        v4f b2 = *(const v4f*)(bp + 8);
        v4f b3 = *(const v4f*)(bp + 12);
        v8h h0, h1;
        #pragma unroll
        for (int i = 0; i < 4; i++) {
          h0[i]     = (_Float16)b0[i];
          h0[i + 4] = (_Float16)b1[i];
          h1[i]     = (_Float16)b2[i];
          h1[i + 4] = (_Float16)b3[i];
        }
        *(v8h*)&Bs[brow][bcoff]     = h0;
        *(v8h*)&Bs[brow][bcoff + 8] = h1;
        if (k0 + TBK < K)
          __builtin_prefetch(bp + (size_t)TBK * N, 0, 1);
      } else {
        #pragma unroll
        for (int j = 0; j < 16; j++) {
          const int gn = n0 + bcoff + j;
          float v = (gn < N) ? Bm[(size_t)gk * N + gn] : 0.f;
          Bs[brow][bcoff + j] = (_Float16)v;
        }
      }
    }
    __syncthreads();

    // ---- fragments per ISA 7.12.2 ----
    // A 16x32 f16: lane%16 = row M; lane half selects K block {0..7,16..23}
    // vs {8..15,24..31}; element j -> k = hlf*8 + (j&7) + (j>>3)*16.
    v16h af[2], bf[2];
    #pragma unroll
    for (int i = 0; i < 2; i++) {
      const int row = wm * 32 + i * 16 + lr;
      #pragma unroll
      for (int j = 0; j < 16; j++) {
        const int kk = (hlf << 3) + (j & 7) + ((j >> 3) << 4);
        af[i][j] = As[row][kk];
      }
    }
    // B 32x16 f16: lane%16 = column N; element j -> k = hlf*8 + (j&7) + (j>>3)*16.
    #pragma unroll
    for (int i = 0; i < 2; i++) {
      const int col0 = wn * 32 + i * 16;
      #pragma unroll
      for (int j = 0; j < 16; j++) bf[i][j] = Bs[(hlf << 3) + (j & 7) + ((j >> 3) << 4)][col0 + lr];
    }

    #pragma unroll
    for (int i = 0; i < 2; i++)
      #pragma unroll
      for (int j = 0; j < 2; j++)
        acc[i][j] = __builtin_amdgcn_wmma_f32_16x16x32_f16(
            false, af[i], false, bf[j], (short)0, acc[i][j], false, false);
    __syncthreads();
  }

  // ---- D layout: element e -> row e + 8*hlf, col lane%16 ----
  #pragma unroll
  for (int i = 0; i < 2; i++)
    #pragma unroll
    for (int j = 0; j < 2; j++)
      #pragma unroll
      for (int e = 0; e < 8; e++) {
        const int gm = m0 + wm * 32 + i * 16 + e + hlf * 8;
        const int gn = n0 + wn * 32 + j * 16 + lr;
        if (gn < N) {
          float z = acc[i][j][e];
          if (bias) z += bias[gn];
          z = apply_epi<EPI>(z);
          if constexpr (ADD) z += addsrc[(size_t)gm * N + gn];
          out[(size_t)gm * N + gn] = z;
        }
      }
}

// --------------------------- reductions ------------------------------------
__device__ __forceinline__ float blk_reduce256(float v, float* red) {
  const int t = threadIdx.x;
  red[t] = v; __syncthreads();
  #pragma unroll
  for (int s = 128; s > 0; s >>= 1) {
    if (t < s) red[t] += red[t + s];
    __syncthreads();
  }
  const float r = red[0]; __syncthreads();
  return r;
}

__device__ __forceinline__ float blk_reduce64(float v, float* red) {
  const int t = threadIdx.x;
  red[t] = v; __syncthreads();
  #pragma unroll
  for (int s = 32; s > 0; s >>= 1) {
    if (t < s) red[t] += red[t + s];
    __syncthreads();
  }
  const float r = red[0]; __syncthreads();
  return r;
}

// -------------------------- small kernels ----------------------------------
// x = LN(emb[idx], ln0)   (one block per token, C=768 = 3 values/thread)
__global__ __launch_bounds__(256)
void embed_ln0_ker(const int* __restrict__ idx, const float* __restrict__ emb,
                   const float* __restrict__ g, const float* __restrict__ b,
                   float* __restrict__ out) {
  __shared__ float red[256];
  const int m = blockIdx.x, t = threadIdx.x;
  const float* e = emb + (size_t)idx[m] * CDIM;
  float v0 = e[t], v1 = e[t + 256], v2 = e[t + 512];
  const float mean = blk_reduce256(v0 + v1 + v2, red) * (1.f / CDIM);
  v0 -= mean; v1 -= mean; v2 -= mean;
  const float var = blk_reduce256(v0 * v0 + v1 * v1 + v2 * v2, red) * (1.f / CDIM);
  const float inv = rsqrtf(var + 1e-5f);
  float* o = out + (size_t)m * CDIM;
  o[t]       = v0 * inv * g[t]       + b[t];
  o[t + 256] = v1 * inv * g[t + 256] + b[t + 256];
  o[t + 512] = v2 * inv * g[t + 512] + b[t + 512];
}

__global__ __launch_bounds__(256)
void ln_ker(const float* __restrict__ x, const float* __restrict__ g,
            const float* __restrict__ b, float* __restrict__ out) {
  __shared__ float red[256];
  const int m = blockIdx.x, t = threadIdx.x;
  const float* xr = x + (size_t)m * CDIM;
  float v0 = xr[t], v1 = xr[t + 256], v2 = xr[t + 512];
  const float mean = blk_reduce256(v0 + v1 + v2, red) * (1.f / CDIM);
  v0 -= mean; v1 -= mean; v2 -= mean;
  const float var = blk_reduce256(v0 * v0 + v1 * v1 + v2 * v2, red) * (1.f / CDIM);
  const float inv = rsqrtf(var + 1e-5f);
  float* o = out + (size_t)m * CDIM;
  o[t]       = v0 * inv * g[t]       + b[t];
  o[t + 256] = v1 * inv * g[t + 256] + b[t + 256];
  o[t + 512] = v2 * inv * g[t + 512] + b[t + 512];
}

// xx = time_shift(src) - src  (zeros at t==0)
__global__ __launch_bounds__(256)
void shiftdiff_ker(const float* __restrict__ src, float* __restrict__ xx) {
  const size_t i = (size_t)blockIdx.x * blockDim.x + threadIdx.x;
  if (i >= (size_t)MTOK * CDIM) return;
  const size_t m = i / CDIM;
  const int t = (int)(m % TSEQ);
  const float prev = (t > 0) ? src[i - CDIM] : 0.f;
  xx[i] = prev - src[i];
}

// per (token, head): v-gating, kk = normalize(k*k_k), k *= 1+(a-1)*k_a
__global__ __launch_bounds__(64)
void tmix_vk_ker(float* __restrict__ k, float* __restrict__ v,
                 float* __restrict__ v_first, float* __restrict__ kk_out,
                 const float* __restrict__ a_sig, const float* __restrict__ vg_sig,
                 const float* __restrict__ k_k, const float* __restrict__ k_a,
                 int layer0) {
  __shared__ float red[64];
  const int mh = blockIdx.x;
  const int m = mh / NH, h = mh % NH;
  const int n = threadIdx.x;
  const int c = h * HS + n;
  const size_t i = (size_t)m * CDIM + c;

  float vv = v[i];
  if (layer0) {
    v_first[i] = vv;
  } else {
    const float s = vg_sig[i];               // sigmoid(v0 + xv@v1@v2)
    vv = vv + (v_first[i] - vv) * s;
    v[i] = vv;
  }
  const float kv = k[i];
  const float kkraw = kv * k_k[c];
  const float ss = blk_reduce64(kkraw * kkraw, red);
  const float nrm = fmaxf(sqrtf(ss), 1e-12f);
  kk_out[i] = kkraw / nrm;
  k[i] = kv * (1.f + (a_sig[i] - 1.f) * k_a[c]);
}

// RWKV7 recurrence: one block per (b,h); thread i owns state row S[i][:].
// w buffer already holds decay = exp(-exp(w)) (fused in GEMM epilogue).
__global__ __launch_bounds__(64)
void rwkv_scan_ker(const float* __restrict__ r, const float* __restrict__ w,
                   const float* __restrict__ k, const float* __restrict__ v,
                   const float* __restrict__ kk, const float* __restrict__ aa,
                   float* __restrict__ y) {
  const int bh = blockIdx.x;
  const int b = bh / NH, h = bh % NH;
  const int i = threadIdx.x;

  float S[HS];
  #pragma unroll
  for (int j = 0; j < HS; j++) S[j] = 0.f;

  __shared__ float rv[HS], wv[HS], kv[HS], av[HS], bv[HS];

  for (int t = 0; t < TSEQ; t++) {
    const size_t base = ((size_t)b * TSEQ + t) * CDIM + h * HS;
    const float kki = kk[base + i];
    rv[i] = r[base + i];
    wv[i] = w[base + i];
    kv[i] = k[base + i];
    av[i] = -kki;                 // a_t = -kk
    bv[i] = kki * aa[base + i];   // b_t = kk * a
    const float vi = v[base + i];
    __syncthreads();

    float sa = 0.f;
    #pragma unroll
    for (int j = 0; j < HS; j++) sa += S[j] * av[j];

    float yi = 0.f;
    #pragma unroll
    for (int j = 0; j < HS; j++) {
      S[j] = S[j] * wv[j] + sa * bv[j] + vi * kv[j];
      yi += S[j] * rv[j];
    }
    y[base + i] = yi;
    __syncthreads();
  }
}

// per (token, head): groupnorm(y) * lnx_g + lnx_b, + (sum r*k*r_k)*v, * g
__global__ __launch_bounds__(64)
void tmix_out_ker(float* __restrict__ y, const float* __restrict__ r,
                  const float* __restrict__ k, const float* __restrict__ v,
                  const float* __restrict__ g, const float* __restrict__ r_k,
                  const float* __restrict__ lnx_g, const float* __restrict__ lnx_b) {
  __shared__ float red[64];
  const int mh = blockIdx.x;
  const int m = mh / NH, h = mh % NH;
  const int n = threadIdx.x;
  const int c = h * HS + n;
  const size_t i = (size_t)m * CDIM + c;

  const float yv = y[i];
  const float mean = blk_reduce64(yv, red) * (1.f / HS);
  const float d = yv - mean;
  const float var = blk_reduce64(d * d, red) * (1.f / HS);
  float gn = d * rsqrtf(var + 0.00064f);
  gn = gn * lnx_g[c] + lnx_b[c];

  const float rksum = blk_reduce64(r[i] * k[i] * r_k[h * HS + n], red);
  y[i] = (gn + rksum * v[i]) * g[i];
}

// ------------------------------ host side ----------------------------------
template<int EPI, bool MIX = false, bool ADD = false>
static void launch_gemm(hipStream_t s, const float* A, const float* AX,
                        const float* mix, const float* Bm, const float* bias,
                        const float* addsrc, float* out, int M, int N, int K) {
  dim3 grid((N + TBN - 1) / TBN, M / TBM);
  gemm_wmma<EPI, MIX, ADD><<<grid, 256, 0, s>>>(A, AX, mix, Bm, bias, addsrc,
                                                out, M, N, K);
}

struct BlkP {
  const float *ln1_g, *ln1_b, *ln2_g, *ln2_b;
  const float *x_r, *x_w, *x_k, *x_v, *x_a, *x_g;
  const float *w0, *w1, *w2, *a0, *a1, *a2, *v0, *v1, *v2, *g1, *g2;
  const float *k_k, *k_a, *r_k;
  const float *Wr, *Wk, *Wv, *Wo;
  const float *lnx_g, *lnx_b;
  const float *fx_k, *Wkey, *Wval;
  const float *ln0_g, *ln0_b;
};

extern "C" void kernel_launch(void* const* d_in, const int* in_sizes, int n_in,
                              void* d_out, int out_size, void* d_ws, size_t ws_size,
                              hipStream_t stream) {
  (void)in_sizes; (void)n_in; (void)out_size; (void)ws_size;
  auto f = [&](int i) { return (const float*)d_in[i]; };

  const int* idx = (const int*)d_in[0];
  int p = 1;
  const float* emb = f(p++);
  BlkP blk[NLAYER];
  for (int i = 0; i < NLAYER; i++) {
    BlkP& B = blk[i];
    B.ln1_g = f(p++); B.ln1_b = f(p++); B.ln2_g = f(p++); B.ln2_b = f(p++);
    B.x_r = f(p++); B.x_w = f(p++); B.x_k = f(p++);
    B.x_v = f(p++); B.x_a = f(p++); B.x_g = f(p++);
    B.w0 = f(p++); B.w1 = f(p++); B.w2 = f(p++);
    B.a0 = f(p++); B.a1 = f(p++); B.a2 = f(p++);
    B.v0 = f(p++); B.v1 = f(p++); B.v2 = f(p++);
    B.g1 = f(p++); B.g2 = f(p++);
    B.k_k = f(p++); B.k_a = f(p++); B.r_k = f(p++);
    B.Wr = f(p++); B.Wk = f(p++); B.Wv = f(p++); B.Wo = f(p++);
    B.lnx_g = f(p++); B.lnx_b = f(p++);
    B.fx_k = f(p++); B.Wkey = f(p++); B.Wval = f(p++);
    if (i == 0) { B.ln0_g = f(p++); B.ln0_b = f(p++); }  // appended last in dict
    else        { B.ln0_g = nullptr; B.ln0_b = nullptr; }
  }
  const float* lno_g = f(p++);
  const float* lno_b = f(p++);
  const float* head  = f(p++);

  // ---- workspace carve-up (fp32) ----
  char* wp = (char*)d_ws;
  auto alloc = [&](size_t nf) { float* q = (float*)wp; wp += nf * sizeof(float); return q; };
  const size_t MC = (size_t)MTOK * CDIM;
  float* X     = alloc(MC);
  float* XLN   = alloc(MC);
  float* XX    = alloc(MC);
  float* Rb    = alloc(MC);
  float* Wb    = alloc(MC);   // decay
  float* Kb    = alloc(MC);
  float* Vb    = alloc(MC);
  float* Ab    = alloc(MC);   // sigmoid gate a
  float* KKb   = alloc(MC);
  float* Gb    = alloc(MC);
  float* VGb   = alloc(MC);   // sigmoid v-gate
  float* VF    = alloc(MC);   // v_first
  float* Yb    = alloc(MC);
  float* HID   = alloc((size_t)MTOK * FFN);
  float* STAGE = alloc((size_t)MTOK * 128);

  const int M = MTOK;
  const size_t nelem = MC;
  const int EW = (int)((nelem + 255) / 256);

  // ---- embed + ln0 ----
  embed_ln0_ker<<<M, 256, 0, stream>>>(idx, emb, blk[0].ln0_g, blk[0].ln0_b, X);

  for (int L = 0; L < NLAYER; L++) {
    const BlkP& B = blk[L];

    // ===== time-mix =====
    ln_ker<<<M, 256, 0, stream>>>(X, B.ln1_g, B.ln1_b, XLN);
    shiftdiff_ker<<<EW, 256, 0, stream>>>(XLN, XX);

    launch_gemm<0, true>(stream, XLN, XX, B.x_r, B.Wr, nullptr, nullptr, Rb, M, CDIM, CDIM);
    launch_gemm<0, true>(stream, XLN, XX, B.x_k, B.Wk, nullptr, nullptr, Kb, M, CDIM, CDIM);
    launch_gemm<0, true>(stream, XLN, XX, B.x_v, B.Wv, nullptr, nullptr, Vb, M, CDIM, CDIM);

    // decay: tanh(xw@w1) @ w2 -> exp(-exp(-softplus(-(w0+.))-0.5))
    launch_gemm<1, true>(stream, XLN, XX, B.x_w, B.w1, nullptr, nullptr, STAGE, M, 64, CDIM);
    launch_gemm<3>(stream, STAGE, nullptr, nullptr, B.w2, B.w0, nullptr, Wb, M, CDIM, 64);

    // a = sigmoid(a0 + xa@a1@a2)
    launch_gemm<0, true>(stream, XLN, XX, B.x_a, B.a1, nullptr, nullptr, STAGE, M, 64, CDIM);
    launch_gemm<2>(stream, STAGE, nullptr, nullptr, B.a2, B.a0, nullptr, Ab, M, CDIM, 64);

    // v-gate = sigmoid(v0 + xv@v1@v2) (layers > 0)
    if (L > 0) {
      launch_gemm<0, true>(stream, XLN, XX, B.x_v, B.v1, nullptr, nullptr, STAGE, M, 32, CDIM);
      launch_gemm<2>(stream, STAGE, nullptr, nullptr, B.v2, B.v0, nullptr, VGb, M, CDIM, 32);
    }

    // g = sigmoid(xg@g1) @ g2
    launch_gemm<2, true>(stream, XLN, XX, B.x_g, B.g1, nullptr, nullptr, STAGE, M, 128, CDIM);
    launch_gemm<0>(stream, STAGE, nullptr, nullptr, B.g2, nullptr, nullptr, Gb, M, CDIM, 128);

    tmix_vk_ker<<<M * NH, 64, 0, stream>>>(Kb, Vb, VF, KKb, Ab, VGb,
                                           B.k_k, B.k_a, L == 0 ? 1 : 0);
    rwkv_scan_ker<<<BBATCH * NH, 64, 0, stream>>>(Rb, Wb, Kb, Vb, KKb, Ab, Yb);
    tmix_out_ker<<<M * NH, 64, 0, stream>>>(Yb, Rb, Kb, Vb, Gb,
                                            B.r_k, B.lnx_g, B.lnx_b);

    // x = x + (y*g) @ Wo   (residual fused into GEMM store)
    launch_gemm<0, false, true>(stream, Yb, nullptr, nullptr, B.Wo, nullptr, X, X, M, CDIM, CDIM);

    // ===== channel-mix =====
    ln_ker<<<M, 256, 0, stream>>>(X, B.ln2_g, B.ln2_b, XLN);
    shiftdiff_ker<<<EW, 256, 0, stream>>>(XLN, XX);
    launch_gemm<4, true>(stream, XLN, XX, B.fx_k, B.Wkey, nullptr, nullptr, HID, M, FFN, CDIM);
    launch_gemm<0, false, true>(stream, HID, nullptr, nullptr, B.Wval, nullptr, X, X, M, CDIM, FFN);
  }

  // ---- final LN + head ----
  ln_ker<<<M, 256, 0, stream>>>(X, lno_g, lno_b, XLN);
  launch_gemm<0>(stream, XLN, nullptr, nullptr, head, nullptr, nullptr,
                 (float*)d_out, M, VOCAB, CDIM);
}
